// Attention_59459527246343
// MI455X (gfx1250) — matrix operand order repaired and checked by reading, not compiled, not run on hardware
//
#include <hip/hip_runtime.h>
#include <hip/hip_bf16.h>

// ---------------------------------------------------------------------------
// Channel-attention (XCA) pipeline for MI455X (gfx1250, wave32, WMMA).
//   B=4, C=128, H=W=256, HEADS=8, ch=16, N=65536.
// Memory-bound overall (~37 GFLOP vs ~0.8 GB traffic with f16 intermediates):
//  K1: qkv = Wqkv @ (s*(x+1))        f16 WMMA GEMM   (128MB rd, 192MB wr)
//  K2: depthwise 3x3 (packed f16) + per-head 16x16 Gram via WMMA + norms
//  K2b: softmax(16) folded with W_proj into per-batch 128x128 matrix M
//  K3: out = M @ v                   f16 WMMA GEMM   (64MB rd, 128MB wr)
// ---------------------------------------------------------------------------

#define N_PIX 65536
#define WPAD  136   // LDS row stride (halfs) for GEMM B-operand staging
#define SPAD  264   // K2 stage row stride (halfs): 4 pad + 256 + 4 pad
#define QK2   272   // K2 q/k Gram staging row stride (halfs)

typedef __attribute__((ext_vector_type(16))) _Float16 v16h;
typedef __attribute__((ext_vector_type(8)))  _Float16 v8h;
typedef __attribute__((ext_vector_type(2)))  _Float16 h2;
typedef __attribute__((ext_vector_type(8)))  float    v8f;

union H16 { v16h v; v8h h[2]; };

static __device__ __forceinline__ v8f wmma_f16(const H16& a, const H16& b, v8f c) {
  // D = A(16x32 f16) * B(32x16 f16) + C(16x16 f32)
  return __builtin_amdgcn_wmma_f32_16x16x32_f16(false, a.v, false, b.v,
                                                (short)0, c, false, false);
}

// ---------------------------------------------------------------------------
// K0: prep. s = p1+p2 ; q_pre = p @ W_kp^T ; W_qkv -> f16 ;
//          W_dw -> duplicated f16-pair table (for scalar-operand pk_fma).
// ---------------------------------------------------------------------------
__global__ __launch_bounds__(256) void k0_prep(
    const float* __restrict__ p, const float* __restrict__ Wqkv,
    const float* __restrict__ Wkp, const float* __restrict__ Wdw,
    float* __restrict__ sbuf, float* __restrict__ qpre,
    _Float16* __restrict__ Wq16, unsigned* __restrict__ Wdwp) {
  const int idx = blockIdx.x * 256 + threadIdx.x;   // 64*256 = 16384 threads
  if (idx < 512) {
    const int b = idx >> 7, c = idx & 127;
    sbuf[idx] = p[b * 256 + c] + p[b * 256 + 128 + c];
    float s = 0.f;
    for (int j = 0; j < 256; ++j) s += p[b * 256 + j] * Wkp[c * 256 + j];
    qpre[idx] = s;
  }
  for (int e = idx; e < 384 * 128; e += 16384) Wq16[e] = (_Float16)Wqkv[e];
  for (int e = idx; e < 384 * 9; e += 16384) {
    const _Float16 wh = (_Float16)Wdw[e];
    const unsigned u = (unsigned)__builtin_bit_cast(unsigned short, wh);
    Wdwp[e] = u | (u << 16);
  }
}

// ---------------------------------------------------------------------------
// K1: qkv[b][384][N] (f16) = W_qkv(384x128) @ (s*(x+1))(128xN), WMMA GEMM.
// Block: one batch x 32 columns.  8 waves x 6 tiles (24 M-tiles x 2 N-tiles).
// ---------------------------------------------------------------------------
__global__ __launch_bounds__(256) void k1_qkv(
    const float* __restrict__ x, const float* __restrict__ sbuf,
    const _Float16* __restrict__ Wq16, _Float16* __restrict__ qkv16) {
  __shared__ _Float16 xs[32 * WPAD];    // B operand: xs[n][k], padded rows
  const int b = blockIdx.x >> 11;
  const int nbase = (blockIdx.x & 2047) * 32;
  const int t = threadIdx.x;
  {
    const int c = t >> 1, jh = t & 1;
    const float s = sbuf[b * 128 + c];
    const float4* xp =
        (const float4*)(x + ((size_t)(b * 128 + c)) * N_PIX + nbase + jh * 16);
#pragma unroll
    for (int i4 = 0; i4 < 4; ++i4) {
      float4 vv = xp[i4];
      const int j = jh * 16 + i4 * 4;
      xs[(j + 0) * WPAD + c] = (_Float16)(s * (vv.x + 1.0f));
      xs[(j + 1) * WPAD + c] = (_Float16)(s * (vv.y + 1.0f));
      xs[(j + 2) * WPAD + c] = (_Float16)(s * (vv.z + 1.0f));
      xs[(j + 3) * WPAD + c] = (_Float16)(s * (vv.w + 1.0f));
    }
  }
  __syncthreads();
  const int wave = t >> 5, lane = t & 31;
  const int loff = lane >> 4, lm = lane & 15;
  for (int tt = 0; tt < 6; ++tt) {
    const int tileid = wave * 6 + tt;
    const int mt = tileid >> 1, nt = tileid & 1;
    v8f acc = {};
    const _Float16* arow = Wq16 + (size_t)(mt * 16 + lm) * 128;
    const _Float16* brow = xs + (nt * 16 + lm) * WPAD;
#pragma unroll
    for (int kk = 0; kk < 128; kk += 32) {
      H16 a, bm;
      a.h[0]  = *(const v8h*)(arow + kk + loff * 8);        // K 0..7 / 8..15
      a.h[1]  = *(const v8h*)(arow + kk + 16 + loff * 8);   // K 16..23 / 24..31
      bm.h[0] = *(const v8h*)(brow + kk + loff * 8);        // K 0..7 / 8..15
      bm.h[1] = *(const v8h*)(brow + kk + 16 + loff * 8);   // K 16..23 / 24..31
      acc = wmma_f16(a, bm, acc);
    }
    _Float16* orow = qkv16 + ((size_t)(b * 384 + mt * 16 + loff * 8)) * N_PIX +
                     nbase + nt * 16 + lm;
#pragma unroll
    for (int r = 0; r < 8; ++r) orow[(size_t)r * N_PIX] = (_Float16)acc[r];
  }
}

// ---------------------------------------------------------------------------
// K2: depthwise 3x3 conv in packed f16 (2 pixels/thread, 2 rows/iter),
// q_pre scale, v (f16) out, per-head norm partials and 16x16 Gram via WMMA.
// Block: (b, head, chunk of 16 rows).  256 threads = 8 waves.
// ---------------------------------------------------------------------------
__global__ __launch_bounds__(256) void k2_dw(
    const _Float16* __restrict__ qkv16, const unsigned* __restrict__ Wdwp,
    const float* __restrict__ qpre, _Float16* __restrict__ vbuf,
    float* __restrict__ Gpart, float* __restrict__ npart) {
  const int blk = blockIdx.x;              // 512 = 4b * 8h * 16 chunks
  const int b = blk >> 7, h = (blk >> 4) & 7, chunk = blk & 15;
  const int t = threadIdx.x, wave = t >> 5, lane = t & 31;
  const int loff = lane >> 4, lm = lane & 15;

  __shared__ _Float16 stage[48 * 4 * SPAD];   // 4 input rows x 48 channels
  __shared__ _Float16 qk[2][2][16 * QK2];     // [q|k][rowsel][ci][px]
  __shared__ float    gred[8 * 256];

  // zero the x-pads once (data region is fully rewritten every iteration)
  for (int e = t; e < 48 * 4; e += 256) {
    *(unsigned long long*)(stage + e * SPAD) = 0ull;         // halfs 0..3
    *(unsigned long long*)(stage + e * SPAD + 260) = 0ull;   // halfs 260..263
  }
  float qp[16];
#pragma unroll
  for (int ci = 0; ci < 16; ++ci) qp[ci] = qpre[b * 128 + h * 16 + ci];
  float sqq[16], sqk[16];
#pragma unroll
  for (int ci = 0; ci < 16; ++ci) { sqq[ci] = 0.f; sqk[ci] = 0.f; }
  v8f accg = {};
  const int xp = (t & 127) * 2;   // pixel-pair base
  const int rsel = t >> 7;        // which of the two output rows
  __syncthreads();

  for (int it = 0; it < 8; ++it) {
    const int y0 = chunk * 16 + it * 2;      // output rows y0, y0+1
    // ---- stage 4 input rows y0-1 .. y0+2 (8B vector loads, zero pad) ----
    for (int i = 0; i < 48; ++i) {
      const int e4 = t + 256 * i;            // group of 4 halfs; 12288 groups
      const int ch = e4 >> 8;                // 256 groups per channel
      const int rem = e4 & 255, r = rem >> 6, x4 = (rem & 63) * 4;
      const int gc = (ch >> 4) * 128 + h * 16 + (ch & 15);
      const int yy = y0 - 1 + r;
      unsigned long long val = 0ull;
      if ((unsigned)yy < 256u)
        val = *(const unsigned long long*)(qkv16 +
              ((size_t)(b * 384 + gc)) * N_PIX + yy * 256 + x4);
      *(unsigned long long*)(stage + (ch * 4 + r) * SPAD + 4 + x4) = val;
    }
    __syncthreads();
    // ---- packed conv: 48 channels, pixels (xp, xp+1) of row y0+rsel ----
#pragma unroll
    for (int part = 0; part < 3; ++part) {
#pragma unroll
      for (int ci = 0; ci < 16; ++ci) {
        const int ch = part * 16 + ci;
        const int gw = (part * 128 + h * 16 + ci) * 9;  // uniform -> s_load
        h2 acc2 = {(_Float16)0, (_Float16)0};
#pragma unroll
        for (int r = 0; r < 3; ++r) {
          const _Float16* base = stage + (ch * 4 + rsel + r) * SPAD + 4 + xp;
          h2 A  = *(const h2*)(base - 2);
          h2 Bv = *(const h2*)(base);
          h2 Cv = *(const h2*)(base + 2);
          h2 pm; pm.x = A.y;  pm.y = Bv.x;
          h2 pp; pp.x = Bv.y; pp.y = Cv.x;
          const h2 w0 = __builtin_bit_cast(h2, Wdwp[gw + r * 3 + 0]);
          const h2 w1 = __builtin_bit_cast(h2, Wdwp[gw + r * 3 + 1]);
          const h2 w2 = __builtin_bit_cast(h2, Wdwp[gw + r * 3 + 2]);
          acc2 = acc2 + pm * w0 + Bv * w1 + pp * w2;   // v_pk_fma_f16
        }
        if (part == 0) {
          const float ax = (float)acc2.x * qp[ci];
          const float ay = (float)acc2.y * qp[ci];
          h2 qv; qv.x = (_Float16)ax; qv.y = (_Float16)ay;
          *(h2*)(&qk[0][rsel][ci * QK2 + xp]) = qv;
          sqq[ci] += ax * ax + ay * ay;
        } else if (part == 1) {
          *(h2*)(&qk[1][rsel][ci * QK2 + xp]) = acc2;
          const float kx = (float)acc2.x, ky = (float)acc2.y;
          sqk[ci] += kx * kx + ky * ky;
        } else {
          *(h2*)(vbuf + ((size_t)(b * 128 + h * 16 + ci)) * N_PIX +
                 (y0 + rsel) * 256 + xp) = acc2;
        }
      }
    }
    __syncthreads();
    // ---- Gram 16x16 += Q(16x32)*K^T(32x16): 16 ksteps, 2 per wave ------
#pragma unroll
    for (int s = 0; s < 2; ++s) {
      const int rs = wave & 1;
      const int pbase = (wave >> 1) * 64 + s * 32;
      const _Float16* qrow = &qk[0][rs][lm * QK2 + pbase];
      const _Float16* krow = &qk[1][rs][lm * QK2 + pbase];
      H16 a, bm;
      a.h[0]  = *(const v8h*)(qrow + loff * 8);
      a.h[1]  = *(const v8h*)(qrow + 16 + loff * 8);
      bm.h[0] = *(const v8h*)(krow + loff * 8);
      bm.h[1] = *(const v8h*)(krow + 16 + loff * 8);
      accg = wmma_f16(a, bm, accg);
    }
    __syncthreads();
  }
  // ---- deterministic partial reductions --------------------------------
#pragma unroll
  for (int r = 0; r < 8; ++r)
    gred[wave * 256 + (r + loff * 8) * 16 + lm] = accg[r];
  float* nb = (float*)stage;   // reuse stage LDS (32*256 floats)
#pragma unroll
  for (int ci = 0; ci < 16; ++ci) {
    nb[ci * 256 + t] = sqq[ci];
    nb[(16 + ci) * 256 + t] = sqk[ci];
  }
  __syncthreads();
  float g = 0.f;
#pragma unroll
  for (int w = 0; w < 8; ++w) g += gred[w * 256 + t];
  Gpart[((size_t)((b * 8 + h) * 16 + chunk)) * 256 + t] = g;
  if (t < 32) {
    float s = 0.f;
    for (int j = 0; j < 256; ++j) s += nb[t * 256 + j];
    npart[((b * 8 + h) * 16 + chunk) * 32 + t] = s;
  }
}

// ---------------------------------------------------------------------------
// K2b: reduce partials -> logits -> softmax(16) -> M = W_proj @ BD(attn), f16
// Block: (b, head).
// ---------------------------------------------------------------------------
__global__ __launch_bounds__(256) void k2b_attn(
    const float* __restrict__ Gpart, const float* __restrict__ npart,
    const float* __restrict__ temp, const float* __restrict__ Wproj,
    _Float16* __restrict__ M16) {
  const int b = blockIdx.x >> 3, h = blockIdx.x & 7;
  const int t = threadIdx.x;
  __shared__ float attn[256];
  __shared__ float nq[16], nk[16];
  float g = 0.f;
  for (int c2 = 0; c2 < 16; ++c2)
    g += Gpart[((size_t)((b * 8 + h) * 16 + c2)) * 256 + t];
  if (t < 32) {
    float s = 0.f;
    for (int c2 = 0; c2 < 16; ++c2)
      s += npart[((b * 8 + h) * 16 + c2) * 32 + t];
    const float n = fmaxf(sqrtf(s), 1e-12f);
    if (t < 16) nq[t] = n; else nk[t - 16] = n;
  }
  __syncthreads();
  const int row = t >> 4, col = t & 15;
  const float logit = g / (nq[row] * nk[col]) * temp[h];
  attn[t] = logit;
  __syncthreads();
  float mx = -1e30f;
  for (int j = 0; j < 16; ++j) mx = fmaxf(mx, attn[row * 16 + j]);
  float sum = 0.f;
  for (int j = 0; j < 16; ++j) sum += __expf(attn[row * 16 + j] - mx);
  const float a = __expf(logit - mx) / sum;
  __syncthreads();
  attn[t] = a;
  __syncthreads();
  // M[o][h*16+d] = sum_ci Wproj[o][h*16+ci] * attn[ci][d]
#pragma unroll
  for (int i = 0; i < 8; ++i) {
    const int u = t + i * 256, o = u >> 4, d = u & 15;
    float m = 0.f;
#pragma unroll
    for (int ci = 0; ci < 16; ++ci)
      m += Wproj[o * 128 + h * 16 + ci] * attn[ci * 16 + d];
    M16[((size_t)(b * 128 + o)) * 128 + h * 16 + d] = (_Float16)m;
  }
}

// ---------------------------------------------------------------------------
// K3: out[b][128][N] (f32) = M_b(128x128 f16) @ v(128xN f16), WMMA GEMM.
// ---------------------------------------------------------------------------
__global__ __launch_bounds__(256) void k3_out(
    const _Float16* __restrict__ vbuf, const _Float16* __restrict__ M16,
    float* __restrict__ out) {
  __shared__ _Float16 xs[32 * WPAD];
  const int b = blockIdx.x >> 11;
  const int nbase = (blockIdx.x & 2047) * 32;
  const int t = threadIdx.x;
  {
    const int c = t >> 1, jh = t & 1;
    const v8h* vp =
        (const v8h*)(vbuf + ((size_t)(b * 128 + c)) * N_PIX + nbase + jh * 16);
    v8h u0 = vp[0], u1 = vp[1];
#pragma unroll
    for (int i = 0; i < 8; ++i) {
      xs[(jh * 16 + i) * WPAD + c]     = u0[i];
      xs[(jh * 16 + 8 + i) * WPAD + c] = u1[i];
    }
  }
  __syncthreads();
  const int wave = t >> 5, lane = t & 31;
  const int loff = lane >> 4, lm = lane & 15;
  for (int tt = 0; tt < 2; ++tt) {
    const int tileid = wave * 2 + tt;
    const int mt = tileid >> 1, nt = tileid & 1;
    v8f acc = {};
    const _Float16* arow = M16 + ((size_t)(b * 128 + mt * 16 + lm)) * 128;
    const _Float16* brow = xs + (nt * 16 + lm) * WPAD;
#pragma unroll
    for (int kk = 0; kk < 128; kk += 32) {
      H16 a, bm;
      a.h[0]  = *(const v8h*)(arow + kk + loff * 8);
      a.h[1]  = *(const v8h*)(arow + kk + 16 + loff * 8);
      bm.h[0] = *(const v8h*)(brow + kk + loff * 8);
      bm.h[1] = *(const v8h*)(brow + kk + 16 + loff * 8);
      acc = wmma_f16(a, bm, acc);
    }
    float* orow = out + ((size_t)(b * 128 + mt * 16 + loff * 8)) * N_PIX +
                  nbase + nt * 16 + lm;
#pragma unroll
    for (int r = 0; r < 8; ++r) orow[(size_t)r * N_PIX] = acc[r];
  }
}

// ---------------------------------------------------------------------------
// Workspace layout (bytes)
// ---------------------------------------------------------------------------
#define OFF_QKV16 ((size_t)0)                    // 4*384*65536*2 = 201326592
#define OFF_VBUF  ((size_t)201326592)            // 4*128*65536*2 =  67108864
#define OFF_WQ16  ((size_t)268435456)            // 384*128*2     =     98304
#define OFF_SBUF  ((size_t)268533760)            // 512*4
#define OFF_QPRE  ((size_t)268535808)            // 512*4
#define OFF_GPART ((size_t)268537856)            // 4*8*16*256*4  =    524288
#define OFF_NPART ((size_t)269062144)            // 4*8*16*32*4   =     65536
#define OFF_M16   ((size_t)269127680)            // 4*128*128*2   =    131072
#define OFF_WDWP  ((size_t)269258752)            // 384*9*4       =     13824

extern "C" void kernel_launch(void* const* d_in, const int* in_sizes, int n_in,
                              void* d_out, int out_size, void* d_ws,
                              size_t ws_size, hipStream_t stream) {
  const float* x     = (const float*)d_in[0];
  const float* p     = (const float*)d_in[1];
  const float* temp  = (const float*)d_in[2];
  const float* Wqkv  = (const float*)d_in[3];
  const float* Wdw   = (const float*)d_in[4];
  const float* Wproj = (const float*)d_in[5];
  const float* Wkp   = (const float*)d_in[6];
  float* out = (float*)d_out;

  char* ws = (char*)d_ws;
  _Float16* qkv16 = (_Float16*)(ws + OFF_QKV16);
  _Float16* vbuf  = (_Float16*)(ws + OFF_VBUF);
  _Float16* Wq16  = (_Float16*)(ws + OFF_WQ16);
  float*    sbuf  = (float*)(ws + OFF_SBUF);
  float*    qpre  = (float*)(ws + OFF_QPRE);
  float*    Gpart = (float*)(ws + OFF_GPART);
  float*    npart = (float*)(ws + OFF_NPART);
  _Float16* M16   = (_Float16*)(ws + OFF_M16);
  unsigned* Wdwp  = (unsigned*)(ws + OFF_WDWP);

  k0_prep<<<64, 256, 0, stream>>>(p, Wqkv, Wkp, Wdw, sbuf, qpre, Wq16, Wdwp);
  k1_qkv<<<8192, 256, 0, stream>>>(x, sbuf, Wq16, qkv16);
  k2_dw<<<512, 256, 0, stream>>>(qkv16, Wdwp, qpre, vbuf, Gpart, npart);
  k2b_attn<<<32, 256, 0, stream>>>(Gpart, npart, temp, Wproj, M16);
  k3_out<<<8192, 256, 0, stream>>>(vbuf, M16, out);
}
